// VulnerabilityGNN_18399639896894
// MI455X (gfx1250) — compile-verified, matrix operand order checked by reading, not run on hardware
//
#include <hip/hip_runtime.h>
#include <hip/hip_bf16.h>

// ---------------------------------------------------------------------------
// VulnerabilityGNN on MI455X (gfx1250, wave32, WMMA)
// Dense-attention softmax algebraically decomposed into colsum + sparse edge
// correction; feature GEMMs run as bf16 WMMA (v_wmma_f32_16x16x32_bf16) with
// a double-buffered GLOBAL_LOAD_ASYNC_TO_LDS_B128 pipeline (ASYNCcnt).
// ---------------------------------------------------------------------------

#define N_NODES 4096
#define N_EDGES 131072
#define HEADS   4
#define DIN     128
#define DH      256
#define HD      (HEADS * DH)     // 1024
#define LRELU   0.2f

typedef __bf16 bf16_t;
typedef __attribute__((ext_vector_type(16))) __bf16 v16bf;
typedef __attribute__((ext_vector_type(8)))  __bf16 v8bf;
typedef __attribute__((ext_vector_type(8)))  float  v8f;

// ------------------------- bf16 WMMA GEMM ----------------------------------
// C[M, ldc] = A[M,K](bf16) x Bt[N,K]^T(bf16)
// Block tile 128x128, 8 waves (4Mx2N), wave tile 32x64, K-step 32.
// Tiles DMA'd straight into LDS with global_load_async_to_lds_b128 and
// double-buffered so the DMA of slab t+1 overlaps the WMMAs of slab t.
#define TM 128
#define TN 128
#define TK 32
#define LDSP 40   // padded bf16 row stride (80B) -> conflict-free ds_load_b128

__global__ __launch_bounds__(256)
void k_gemm_bf16_tn(const bf16_t* __restrict__ A,
                    const bf16_t* __restrict__ Bt,
                    float* __restrict__ C,
                    int K, int ldc)
{
    __shared__ __align__(16) bf16_t sA[2][TM * LDSP];
    __shared__ __align__(16) bf16_t sB[2][TN * LDSP];

    const int tid  = threadIdx.x;
    const int lane = tid & 31;
    const int wm   = (tid >> 5) >> 1;   // 0..3
    const int wn   = (tid >> 5) & 1;    // 0..1
    const long m0  = (long)blockIdx.x * TM;
    const long n0  = (long)blockIdx.y * TN;

    // cooperative tile load mapping: 512 chunks of 8 bf16, 2 per thread/tile
    const int c0  = tid * 2;
    const int lr0 = c0 >> 2;        const int lk0 = (c0 & 3) * 8;
    const int lr1 = (c0 + 1) >> 2;  const int lk1 = ((c0 + 1) & 3) * 8;

    // per-lane LDS byte addresses (low 32 bits of generic pointer = LDS offset)
    unsigned dA0[2], dA1[2], dB0[2], dB1[2];
    #pragma unroll
    for (int b = 0; b < 2; ++b) {
        dA0[b] = (unsigned)(uintptr_t)(&sA[b][lr0 * LDSP + lk0]);
        dA1[b] = (unsigned)(uintptr_t)(&sA[b][lr1 * LDSP + lk1]);
        dB0[b] = (unsigned)(uintptr_t)(&sB[b][lr0 * LDSP + lk0]);
        dB1[b] = (unsigned)(uintptr_t)(&sB[b][lr1 * LDSP + lk1]);
    }

    auto issue = [&](int k0, int b) {
        const bf16_t* a0 = A  + (m0 + lr0) * K + k0 + lk0;
        const bf16_t* a1 = A  + (m0 + lr1) * K + k0 + lk1;
        const bf16_t* p0 = Bt + (n0 + lr0) * K + k0 + lk0;
        const bf16_t* p1 = Bt + (n0 + lr1) * K + k0 + lk1;
        asm volatile(
            "global_load_async_to_lds_b128 %0, %4, off\n\t"
            "global_load_async_to_lds_b128 %1, %5, off\n\t"
            "global_load_async_to_lds_b128 %2, %6, off\n\t"
            "global_load_async_to_lds_b128 %3, %7, off"
            :: "v"(dA0[b]), "v"(dA1[b]), "v"(dB0[b]), "v"(dB1[b]),
               "v"(a0), "v"(a1), "v"(p0), "v"(p1)
            : "memory");
    };

    v8f acc[2][4] = {};
    const int rlo   = lane & 15;        // row-within-16 for fragments
    const int khalf = (lane >> 4) * 8;  // K sub-offset per ISA A/B layout

    issue(0, 0);
    const int nT = K / TK;
    for (int t = 0; t < nT; ++t) {
        const int cur = t & 1;
        asm volatile("s_wait_asynccnt 0" ::: "memory");  // my DMA done
        __syncthreads();                                  // everyone's DMA done
        if (t + 1 < nT) issue((t + 1) * TK, cur ^ 1);     // overlap next slab

        v16bf aF[2], bF[4];
        #pragma unroll
        for (int mi = 0; mi < 2; ++mi) {
            const bf16_t* p = &sA[cur][(wm * 32 + mi * 16 + rlo) * LDSP + khalf];
            v8bf lo = *(const v8bf*)p;
            v8bf hi = *(const v8bf*)(p + 16);
            aF[mi] = __builtin_shufflevector(lo, hi, 0,1,2,3,4,5,6,7,8,9,10,11,12,13,14,15);
        }
        #pragma unroll
        for (int ni = 0; ni < 4; ++ni) {
            const bf16_t* p = &sB[cur][(wn * 64 + ni * 16 + rlo) * LDSP + khalf];
            v8bf lo = *(const v8bf*)p;
            v8bf hi = *(const v8bf*)(p + 16);
            bF[ni] = __builtin_shufflevector(lo, hi, 0,1,2,3,4,5,6,7,8,9,10,11,12,13,14,15);
        }
        #pragma unroll
        for (int mi = 0; mi < 2; ++mi)
            #pragma unroll
            for (int ni = 0; ni < 4; ++ni)
                acc[mi][ni] = __builtin_amdgcn_wmma_f32_16x16x32_bf16(
                    false, aF[mi], false, bF[ni], (short)0, acc[mi][ni], false, false);
    }

    #pragma unroll
    for (int mi = 0; mi < 2; ++mi) {
        #pragma unroll
        for (int ni = 0; ni < 4; ++ni) {
            const long cn = n0 + wn * 64 + ni * 16 + rlo;
            const long cm = m0 + wm * 32 + mi * 16 + (lane >> 4) * 8;
            #pragma unroll
            for (int r = 0; r < 8; ++r)
                C[(cm + r) * ldc + cn] = acc[mi][ni][r];
        }
    }
}

// ------------------------- small utility kernels ---------------------------
__global__ void k_cvt_bf16(const float* __restrict__ in, bf16_t* __restrict__ out, int n) {
    int i = blockIdx.x * blockDim.x + threadIdx.x;
    if (i < n) out[i] = (bf16_t)in[i];
}

// in[R,C] f32 -> out[C,R] bf16 (transposed for TN GEMM B operand)
__global__ void k_cvt_transpose(const float* __restrict__ in, bf16_t* __restrict__ out,
                                int R, int C) {
    int idx = blockIdx.x * blockDim.x + threadIdx.x;
    if (idx >= R * C) return;
    int o = idx / R;
    int i = idx - o * R;
    out[idx] = (bf16_t)in[i * C + o];
}

__global__ void k_fill(float* __restrict__ p, float v, int n) {
    int i = blockIdx.x * blockDim.x + threadIdx.x;
    if (i < n) p[i] = v;
}

__global__ void k_proj_epilogue(const float* __restrict__ C, const float* __restrict__ bp,
                                bf16_t* __restrict__ hb) {
    int idx = blockIdx.x * blockDim.x + threadIdx.x;
    if (idx >= N_NODES * DH) return;
    float v = C[idx] + bp[idx & (DH - 1)];
    hb[idx] = (bf16_t)fmaxf(v, 0.0f);
}

// e_src/e_dst = a . Wh  per (head, node)
__global__ void k_escore(const float* __restrict__ Wh, const float* __restrict__ a,
                         float* __restrict__ esrc, float* __restrict__ edst) {
    int idx = blockIdx.x * blockDim.x + threadIdx.x;
    if (idx >= HEADS * N_NODES) return;
    int h = idx >> 12;
    const float* w  = Wh + (long)idx * DH;
    const float* as = a + h * 2 * DH;
    const float* ad = as + DH;
    float s0 = 0.f, s1 = 0.f;
    #pragma unroll 4
    for (int o = 0; o < DH; ++o) { float x = w[o]; s0 += as[o] * x; s1 += ad[o] * x; }
    esrc[idx] = s0; edst[idx] = s1;
}

// colsum[h][o] += partial sum over a 256-node slab (grid.y slabs)
__global__ void k_colsum(const float* __restrict__ Wh, float* __restrict__ colsum) {
    int idx = blockIdx.x * blockDim.x + threadIdx.x;  // h*256+o over HEADS*DH
    int h = idx >> 8, o = idx & 255;
    const float* p = Wh + ((long)h * N_NODES) * DH + o;
    int n0 = blockIdx.y * 256;
    float s = 0.f;
    for (int n = n0; n < n0 + 256; ++n) s += p[(long)n * DH];
    atomicAdd(&colsum[idx], s);
}

// per (head,edge): w-1 = exp(leaky(e_src+e_dst))-1 ; Z[src] += w-1
__global__ void k_edge_w(const int* __restrict__ ei, const float* __restrict__ esrc,
                         const float* __restrict__ edst, float* __restrict__ wm1,
                         float* __restrict__ Z) {
    int idx = blockIdx.x * blockDim.x + threadIdx.x;
    if (idx >= HEADS * N_EDGES) return;
    int h = idx >> 17, e = idx & (N_EDGES - 1);
    int s = ei[e], d = ei[N_EDGES + e];
    float z = esrc[(h << 12) + s] + edst[(h << 12) + d];
    z = (z > 0.f) ? z : LRELU * z;
    float w = __expf(z) - 1.0f;
    wm1[idx] = w;
    atomicAdd(&Z[(h << 12) + s], w);
}

// one wave per (head,edge): acc[h][src][:] += (w-1) * Wh[h][dst][:]
__global__ __launch_bounds__(256)
void k_edge_agg(const int* __restrict__ ei, const float* __restrict__ wm1,
                const float* __restrict__ Wh, float* __restrict__ acc) {
    int wid  = blockIdx.x * 8 + (threadIdx.x >> 5);
    int lane = threadIdx.x & 31;
    if (wid >= HEADS * N_EDGES) return;
    int h = wid & 3, e = wid >> 2;
    int s = ei[e], d = ei[N_EDGES + e];
    float c = wm1[(h << 17) + e];
    const float4* src = (const float4*)(Wh + ((long)(h << 12) + d) * DH) + lane * 2;
    float* dst = acc + ((long)(h << 12) + s) * DH + lane * 8;
    float4 v0 = src[0], v1 = src[1];
    atomicAdd(dst + 0, c * v0.x); atomicAdd(dst + 1, c * v0.y);
    atomicAdd(dst + 2, c * v0.z); atomicAdd(dst + 3, c * v0.w);
    atomicAdd(dst + 4, c * v1.x); atomicAdd(dst + 5, c * v1.y);
    atomicAdd(dst + 6, c * v1.z); atomicAdd(dst + 7, c * v1.w);
}

__global__ void k_invz(const float* __restrict__ Z, float* __restrict__ Sinv) {
    __shared__ float red[256];
    int h = blockIdx.x, tid = threadIdx.x;
    float s = 0.f;
    for (int n = tid; n < N_NODES; n += 256) s += 1.0f / Z[(h << 12) + n];
    red[tid] = s; __syncthreads();
    for (int o = 128; o; o >>= 1) { if (tid < o) red[tid] += red[tid + o]; __syncthreads(); }
    if (tid == 0) Sinv[h] = red[0];
}

__global__ void k_edge_imp(const int* __restrict__ ei, const float* __restrict__ wm1,
                           const float* __restrict__ Z, float* __restrict__ imp) {
    int idx = blockIdx.x * blockDim.x + threadIdx.x;
    if (idx >= HEADS * N_EDGES) return;
    int h = idx >> 17, e = idx & (N_EDGES - 1);
    int s = ei[e], d = ei[N_EDGES + e];
    atomicAdd(&imp[d], wm1[idx] / Z[(h << 12) + s]);
}

__global__ void k_imp_base(float* __restrict__ imp, const float* __restrict__ Sinv) {
    int n = blockIdx.x * blockDim.x + threadIdx.x;
    if (n < N_NODES) imp[n] += Sinv[0] + Sinv[1] + Sinv[2] + Sinv[3];
}

// h' = elu((colsum + acc)/Z), write bf16 (next layer A) + f32 (pooling)
__global__ void k_layer_epilogue(const float* __restrict__ colsum, const float* __restrict__ acc,
                                 const float* __restrict__ Z, bf16_t* __restrict__ hb,
                                 float* __restrict__ hf) {
    long idx = (long)blockIdx.x * blockDim.x + threadIdx.x;
    if (idx >= (long)N_NODES * HD) return;
    int n = (int)(idx >> 10), c = (int)(idx & 1023);
    int h = c >> 8, o = c & 255;
    float v = (colsum[(h << 8) + o] + acc[((long)(h << 12) + n) * DH + o]) / Z[(h << 12) + n];
    float r = v > 0.f ? v : (__expf(v) - 1.0f);
    hb[idx] = (bf16_t)r;
    hf[idx] = r;
}

__global__ void k_pool_score(const float* __restrict__ hf, const float* __restrict__ pw,
                             const float* __restrict__ pb, float* __restrict__ sbuf) {
    int n = blockIdx.x * blockDim.x + threadIdx.x;
    if (n >= N_NODES) return;
    const float* row = hf + (long)n * HD;
    float s = pb[0];
    #pragma unroll 4
    for (int j = 0; j < HD; ++j) s += row[j] * pw[j];
    sbuf[n] = s;
}

__global__ __launch_bounds__(1024)
void k_pool_emb(const float* __restrict__ s, const float* __restrict__ hf,
                float* __restrict__ emb) {
    __shared__ float sp[N_NODES];
    __shared__ float red[1024];
    int tid = threadIdx.x;
    float mx = -3.4e38f;
    for (int n = tid; n < N_NODES; n += 1024) mx = fmaxf(mx, s[n]);
    red[tid] = mx; __syncthreads();
    for (int o = 512; o; o >>= 1) { if (tid < o) red[tid] = fmaxf(red[tid], red[tid + o]); __syncthreads(); }
    mx = red[0]; __syncthreads();
    float sum = 0.f;
    for (int n = tid; n < N_NODES; n += 1024) { float e = __expf(s[n] - mx); sp[n] = e; sum += e; }
    red[tid] = sum; __syncthreads();
    for (int o = 512; o; o >>= 1) { if (tid < o) red[tid] += red[tid + o]; __syncthreads(); }
    float inv = 1.0f / red[0];
    __syncthreads();
    float a = 0.f;
    for (int n = 0; n < N_NODES; ++n) a += sp[n] * hf[(long)n * HD + tid];
    emb[tid] = a * inv;
}

__global__ void k_classifier(const float* __restrict__ emb,
                             const float* __restrict__ w1, const float* __restrict__ b1,
                             const float* __restrict__ w2, const float* __restrict__ b2,
                             float* __restrict__ out) {
    __shared__ float t1[DH];
    __shared__ float lg[8];
    int tid = threadIdx.x;
    float s = b1[tid];
    for (int j = 0; j < HD; ++j) s += emb[j] * w1[j * DH + tid];
    t1[tid] = fmaxf(s, 0.f);
    __syncthreads();
    if (tid < 7) {
        float l = b2[tid];
        for (int j = 0; j < DH; ++j) l += t1[j] * w2[j * 7 + tid];
        lg[tid] = l;
    }
    __syncthreads();
    if (tid == 0) {
        float m = lg[0];
        for (int j = 1; j < 7; ++j) m = fmaxf(m, lg[j]);
        float den = 0.f, e[7];
        for (int j = 0; j < 7; ++j) { e[j] = __expf(lg[j] - m); den += e[j]; }
        for (int j = 0; j < 7; ++j) out[j] = e[j] / den;
    }
}

__global__ void k_final_imp(const float* __restrict__ imp, float* __restrict__ out) {
    int n = blockIdx.x * blockDim.x + threadIdx.x;
    if (n < N_NODES) out[n] = imp[n] * (1.0f / 12.0f);   // /(3*H)
}

// ---------------------------------------------------------------------------
extern "C" void kernel_launch(void* const* d_in, const int* in_sizes, int n_in,
                              void* d_out, int out_size, void* d_ws, size_t ws_size,
                              hipStream_t stream) {
    const float* x      = (const float*)d_in[0];
    const int*   ei     = (const int*)d_in[1];
    const float* Wp     = (const float*)d_in[2];
    const float* bp     = (const float*)d_in[3];
    const float* Wl[3]  = {(const float*)d_in[4], (const float*)d_in[6], (const float*)d_in[8]};
    const float* al[3]  = {(const float*)d_in[5], (const float*)d_in[7], (const float*)d_in[9]};
    const float* pool_w = (const float*)d_in[10];
    const float* pool_b = (const float*)d_in[11];
    const float* c1_w   = (const float*)d_in[12];
    const float* c1_b   = (const float*)d_in[13];
    const float* c2_w   = (const float*)d_in[14];
    const float* c2_b   = (const float*)d_in[15];
    float* out = (float*)d_out;

    char* ws = (char*)d_ws;
    size_t off = 0;
    auto alloc = [&](size_t bytes) -> char* {
        char* p = ws + off; off += (bytes + 255) & ~(size_t)255; return p;
    };
    bf16_t* xb    = (bf16_t*)alloc((size_t)N_NODES * DIN * 2);
    bf16_t* wt    = (bf16_t*)alloc((size_t)HEADS * HD * DH * 2);
    bf16_t* hbA   = (bf16_t*)alloc((size_t)N_NODES * HD * 2);
    bf16_t* hbB   = (bf16_t*)alloc((size_t)N_NODES * HD * 2);
    float*  hf    = (float*) alloc((size_t)N_NODES * HD * 4);
    float*  Whf   = (float*) alloc((size_t)HEADS * N_NODES * DH * 4);
    float*  accb  = (float*) alloc((size_t)HEADS * N_NODES * DH * 4);
    float*  esrc  = (float*) alloc((size_t)HEADS * N_NODES * 4);
    float*  edst  = (float*) alloc((size_t)HEADS * N_NODES * 4);
    float*  Zb    = (float*) alloc((size_t)HEADS * N_NODES * 4);
    float*  wm1b  = (float*) alloc((size_t)HEADS * N_EDGES * 4);
    float*  csb   = (float*) alloc((size_t)HEADS * DH * 4);
    float*  impb  = (float*) alloc((size_t)N_NODES * 4);
    float*  Sinvb = (float*) alloc(64);
    float*  sbuf  = (float*) alloc((size_t)N_NODES * 4);
    float*  emb   = (float*) alloc((size_t)HD * 4);

    const int T = 256;
    #define GRID1(n) dim3((unsigned)(((n) + T - 1) / T))

    // projection: h = relu(x @ Wp + bp)  via bf16 WMMA
    k_cvt_bf16<<<GRID1(N_NODES * DIN), T, 0, stream>>>(x, xb, N_NODES * DIN);
    k_cvt_transpose<<<GRID1(DIN * DH), T, 0, stream>>>(Wp, wt, DIN, DH);
    k_gemm_bf16_tn<<<dim3(N_NODES / TM, DH / TN), 256, 0, stream>>>(xb, wt, Whf, DIN, DH);
    k_proj_epilogue<<<GRID1(N_NODES * DH), T, 0, stream>>>(Whf, bp, hbA);
    k_fill<<<GRID1(N_NODES), T, 0, stream>>>(impb, 0.f, N_NODES);

    const bf16_t* hcur = hbA;
    int K = DH;
    for (int l = 0; l < 3; ++l) {
        const int F = K;  // input features this layer (256 then 1024)
        for (int h = 0; h < HEADS; ++h)
            k_cvt_transpose<<<GRID1(F * DH), T, 0, stream>>>(
                Wl[l] + (size_t)h * F * DH, wt + (size_t)h * DH * F, F, DH);
        for (int h = 0; h < HEADS; ++h)
            k_gemm_bf16_tn<<<dim3(N_NODES / TM, DH / TN), 256, 0, stream>>>(
                hcur, wt + (size_t)h * DH * F, Whf + (size_t)h * N_NODES * DH, F, DH);

        k_escore<<<GRID1(HEADS * N_NODES), T, 0, stream>>>(Whf, al[l], esrc, edst);
        k_fill<<<GRID1(HEADS * DH), T, 0, stream>>>(csb, 0.f, HEADS * DH);
        k_colsum<<<dim3(HEADS * DH / T, N_NODES / 256), T, 0, stream>>>(Whf, csb);
        k_fill<<<GRID1(HEADS * N_NODES), T, 0, stream>>>(Zb, (float)N_NODES, HEADS * N_NODES);
        k_fill<<<GRID1(HEADS * N_NODES * DH), T, 0, stream>>>(accb, 0.f, HEADS * N_NODES * DH);
        k_edge_w<<<GRID1(HEADS * N_EDGES), T, 0, stream>>>(ei, esrc, edst, wm1b, Zb);
        k_edge_agg<<<dim3(HEADS * N_EDGES / 8), 256, 0, stream>>>(ei, wm1b, Whf, accb);
        k_invz<<<dim3(HEADS), 256, 0, stream>>>(Zb, Sinvb);
        k_edge_imp<<<GRID1(HEADS * N_EDGES), T, 0, stream>>>(ei, wm1b, Zb, impb);
        k_imp_base<<<GRID1(N_NODES), T, 0, stream>>>(impb, Sinvb);

        bf16_t* hnext = (l & 1) ? hbA : hbB;
        k_layer_epilogue<<<GRID1(N_NODES * HD), T, 0, stream>>>(csb, accb, Zb, hnext, hf);
        hcur = hnext;
        K = HD;
    }

    k_pool_score<<<GRID1(N_NODES), T, 0, stream>>>(hf, pool_w, pool_b, sbuf);
    k_pool_emb<<<dim3(1), 1024, 0, stream>>>(sbuf, hf, emb);
    k_classifier<<<dim3(1), 256, 0, stream>>>(emb, c1_w, c1_b, c2_w, c2_b, out);
    k_final_imp<<<GRID1(N_NODES), T, 0, stream>>>(impb, out + 7);

    (void)in_sizes; (void)n_in; (void)out_size; (void)ws_size;
    #undef GRID1
}
